// general_attn_12584254178007
// MI455X (gfx1250) — matrix operand order repaired and checked by reading, not compiled, not run on hardware
//
#include <hip/hip_runtime.h>

#define DIM    1024
#define HEADS  16
#define HD     64
#define BATCH  4
#define SEQ    2048
#define MTOT   (BATCH*SEQ)   // 8192 rows
#define QKVN   (3*DIM)       // 3072
#define LDST   72            // 64 + 8 bf16 pad -> 144B row, 16B aligned

typedef __bf16 bf16_t;
typedef __attribute__((ext_vector_type(16))) __bf16 v16bf;
typedef __attribute__((ext_vector_type(8)))  __bf16 v8bf;
typedef __attribute__((ext_vector_type(8)))  float  v8f;

union Frag {
    v16bf v;
    v8bf  h[2];
};

__device__ __forceinline__ v8f wmma_bf16(const Frag& a, const Frag& b, v8f c) {
    return __builtin_amdgcn_wmma_f32_16x16x32_bf16(
        false, a.v, false, b.v, (short)0, c, false, false);
}

// CDNA5 async global->LDS copy, 16B per lane, tracked by ASYNCcnt.
// Generic pointers to __shared__ carry the wave-relative LDS offset in the
// low 32 bits (ISA 10.2).
__device__ __forceinline__ void async_g2l_b128(void* lds, const void* gmem) {
    uint32_t l = (uint32_t)(uintptr_t)lds;
    asm volatile("global_load_async_to_lds_b128 %0, %1, off"
                 :: "v"(l), "v"(gmem) : "memory");
}
__device__ __forceinline__ void wait_async0() {
    asm volatile("s_wait_asynccnt 0x0" ::: "memory");
}

// ---------------------------------------------------------------- convert
__global__ void cvt_f32_bf16(const float* __restrict__ src,
                             bf16_t* __restrict__ dst, int n) {
    int i = blockIdx.x * blockDim.x + threadIdx.x;
    if (i < n) dst[i] = (bf16_t)src[i];
}

// ---------------------------------------------------------------- GEMM core
// 128x128 block tile, KT=64, 8 waves (2x4), double-buffered async staging.
__device__ __forceinline__ void gemm_stage(const bf16_t* ag, const bf16_t* bg,
                                           bf16_t* al, bf16_t* bl) {
    #pragma unroll
    for (int u = 0; u < 4; u++) {
        async_g2l_b128(al + u * 8, ag + u * 8);
        async_g2l_b128(bl + u * 8, bg + u * 8);
    }
}

__device__ __forceinline__ void gemm_compute(const bf16_t* As, const bf16_t* Bs,
                                             int wm, int wn, int l16, int lh,
                                             v8f acc[4][2]) {
    #pragma unroll
    for (int ks = 0; ks < 64; ks += 32) {
        Frag a[4], b[2];
        #pragma unroll
        for (int tm = 0; tm < 4; tm++) {
            int r = wm * 64 + tm * 16 + l16;
            a[tm].h[0] = *(const v8bf*)&As[r * LDST + ks + lh * 8];
            a[tm].h[1] = *(const v8bf*)&As[r * LDST + ks + lh * 8 + 16];
        }
        #pragma unroll
        for (int tn = 0; tn < 2; tn++) {
            int c = wn * 32 + tn * 16 + l16;
            b[tn].h[0] = *(const v8bf*)&Bs[c * LDST + ks + lh * 8];
            b[tn].h[1] = *(const v8bf*)&Bs[c * LDST + ks + lh * 8 + 16];
        }
        #pragma unroll
        for (int tm = 0; tm < 4; tm++)
            #pragma unroll
            for (int tn = 0; tn < 2; tn++)
                acc[tm][tn] = wmma_bf16(a[tm], b[tn], acc[tm][tn]);
    }
}

// Computes acc = A[128 rows @m0] x W[128 rows @n0]^T over K=1024.
__device__ __forceinline__ void gemm_core(const bf16_t* __restrict__ Aptr,
                                          const bf16_t* __restrict__ Wptr,
                                          v8f acc[4][2])
{
    __shared__ __attribute__((aligned(16))) bf16_t As[2][128 * LDST];
    __shared__ __attribute__((aligned(16))) bf16_t Bs[2][128 * LDST];

    const int tid  = threadIdx.x;
    const int wid  = tid >> 5, lane = tid & 31;
    const int l16  = lane & 15, lh = lane >> 4;
    const int wm   = wid >> 2,  wn = wid & 3;
    const int m0   = blockIdx.x * 128, n0 = blockIdx.y * 128;
    const int lrow = tid >> 1,  lseg = tid & 1;

    const bf16_t* arow = Aptr + (size_t)(m0 + lrow) * DIM + lseg * 32;
    const bf16_t* brow = Wptr + (size_t)(n0 + lrow) * DIM + lseg * 32;
    bf16_t* al0 = &As[0][lrow * LDST + lseg * 32];
    bf16_t* bl0 = &Bs[0][lrow * LDST + lseg * 32];
    bf16_t* al1 = &As[1][lrow * LDST + lseg * 32];
    bf16_t* bl1 = &Bs[1][lrow * LDST + lseg * 32];

    #pragma unroll
    for (int a = 0; a < 4; a++)
        #pragma unroll
        for (int b = 0; b < 2; b++)
            #pragma unroll
            for (int j = 0; j < 8; j++) acc[a][b][j] = 0.f;

    gemm_stage(arow, brow, al0, bl0);                    // prologue: tile 0

    for (int kt = 0; kt < DIM; kt += 128) {
        wait_async0();
        __syncthreads();                                  // buffer 0 ready
        gemm_stage(arow + kt + 64, brow + kt + 64, al1, bl1);   // prefetch
        gemm_compute(&As[0][0], &Bs[0][0], wm, wn, l16, lh, acc);

        wait_async0();
        __syncthreads();                                  // buffer 1 ready
        if (kt + 128 < DIM)
            gemm_stage(arow + kt + 128, brow + kt + 128, al0, bl0);
        gemm_compute(&As[1][0], &Bs[1][0], wm, wn, l16, lh, acc);
    }
}

// ---------------------------------------------------------------- QKV GEMM
__global__ __launch_bounds__(256)
void gemm_qkv(const bf16_t* __restrict__ X, const bf16_t* __restrict__ W,
              const float* __restrict__ bias,
              bf16_t* __restrict__ Qo, bf16_t* __restrict__ Ko,
              bf16_t* __restrict__ Vt)
{
    v8f acc[4][2];
    gemm_core(X, W, acc);

    const int tid = threadIdx.x;
    const int wid = tid >> 5, lane = tid & 31;
    const int l16 = lane & 15, lh = lane >> 4;
    const int wm  = wid >> 2,  wn = wid & 3;
    const int m0  = blockIdx.x * 128, n0 = blockIdx.y * 128;

    #pragma unroll
    for (int tn = 0; tn < 2; tn++) {
        int gcol  = n0 + wn * 32 + tn * 16 + l16;
        float bv  = bias[gcol];
        int which = gcol >> 10;          // 0=q 1=k 2=v
        int hh    = (gcol >> 6) & 15;
        int dd    = gcol & 63;
        #pragma unroll
        for (int tm = 0; tm < 4; tm++) {
            #pragma unroll
            for (int j = 0; j < 8; j++) {
                int grow = m0 + wm * 64 + tm * 16 + lh * 8 + j;
                int bb = grow >> 11, nn = grow & 2047;
                bf16_t val = (bf16_t)(acc[tm][tn][j] + bv);
                if (which == 2)
                    Vt[(size_t)((bb * HEADS + hh) * HD + dd) * SEQ + nn] = val;
                else if (which == 1)
                    Ko[(size_t)((bb * HEADS + hh) * SEQ + nn) * HD + dd] = val;
                else
                    Qo[(size_t)((bb * HEADS + hh) * SEQ + nn) * HD + dd] = val;
            }
        }
    }
}

// ---------------------------------------------------------------- out GEMM
__global__ __launch_bounds__(256)
void gemm_out(const bf16_t* __restrict__ A, const bf16_t* __restrict__ W,
              const float* __restrict__ bias, float* __restrict__ Out)
{
    v8f acc[4][2];
    gemm_core(A, W, acc);

    const int tid = threadIdx.x;
    const int wid = tid >> 5, lane = tid & 31;
    const int l16 = lane & 15, lh = lane >> 4;
    const int wm  = wid >> 2,  wn = wid & 3;
    const int m0  = blockIdx.x * 128, n0 = blockIdx.y * 128;

    #pragma unroll
    for (int tn = 0; tn < 2; tn++) {
        int gcol = n0 + wn * 32 + tn * 16 + l16;
        float bv = bias[gcol];
        #pragma unroll
        for (int tm = 0; tm < 4; tm++) {
            #pragma unroll
            for (int j = 0; j < 8; j++) {
                int grow = m0 + wm * 64 + tm * 16 + lh * 8 + j;
                Out[(size_t)grow * DIM + gcol] = acc[tm][tn][j] + bv;
            }
        }
    }
}

// ---------------------------------------------------------------- flash attn
// grid: (SEQ/128, B*H). 8 waves/block, each wave owns 16 query rows.
// K/V tiles double-buffered via async copies; P re-staged per wave in LDS.
__global__ __launch_bounds__(256)
void flash_attn(const bf16_t* __restrict__ Q, const bf16_t* __restrict__ K,
                const bf16_t* __restrict__ Vt, bf16_t* __restrict__ Ob)
{
    __shared__ __attribute__((aligned(16))) bf16_t Plds[8][16 * HD]; // 16KB
    __shared__ __attribute__((aligned(16))) bf16_t Ks[2][64 * LDST]; // 18KB
    __shared__ __attribute__((aligned(16))) bf16_t Vs[2][64 * LDST]; // 18KB

    const int tid = threadIdx.x;
    const int wid = tid >> 5, lane = tid & 31;
    const int l16 = lane & 15, lh = lane >> 4;
    const int bh  = blockIdx.y;
    const int q0  = blockIdx.x * 128 + wid * 16;
    const int srow = tid >> 2, sseg = tid & 3;   // staging split: 64 rows x 4

    const bf16_t* Qh = Q  + (size_t)bh * SEQ * HD;
    const bf16_t* Kh = K  + (size_t)bh * SEQ * HD;
    const bf16_t* Vh = Vt + (size_t)bh * HD * SEQ;
    bf16_t* Pw = &Plds[wid][0];

    // Q fragments for this wave's 16 rows (reused across all key tiles)
    Frag qa[2];
    #pragma unroll
    for (int ki = 0; ki < 2; ki++) {
        const bf16_t* p = Qh + (size_t)(q0 + l16) * HD + ki * 32 + lh * 8;
        qa[ki].h[0] = *(const v8bf*)(p);
        qa[ki].h[1] = *(const v8bf*)(p + 16);
    }

    v8f o[4];
    float rmax[8], rsum[8];
    #pragma unroll
    for (int j = 0; j < 8; j++) { rmax[j] = -1e30f; rsum[j] = 0.f; }
    #pragma unroll
    for (int d = 0; d < 4; d++)
        #pragma unroll
        for (int j = 0; j < 8; j++) o[d][j] = 0.f;

    auto stage_kv = [&](int buf, int kv) {
        const bf16_t* kg = Kh + (size_t)(kv + srow) * HD + sseg * 16;
        const bf16_t* vg = Vh + (size_t)srow * SEQ + kv + sseg * 16;
        bf16_t* kl = &Ks[buf][srow * LDST + sseg * 16];
        bf16_t* vl = &Vs[buf][srow * LDST + sseg * 16];
        async_g2l_b128(kl,     kg);
        async_g2l_b128(kl + 8, kg + 8);
        async_g2l_b128(vl,     vg);
        async_g2l_b128(vl + 8, vg + 8);
    };

    auto kv_body = [&](const bf16_t* Ksb, const bf16_t* Vsb) {
        // ---- S = Q @ K^T  (4 tiles of 16 keys, gemm-K = hd = 64)
        v8f s[4];
        #pragma unroll
        for (int tn = 0; tn < 4; tn++)
            #pragma unroll
            for (int j = 0; j < 8; j++) s[tn][j] = 0.f;

        #pragma unroll
        for (int tn = 0; tn < 4; tn++) {
            int krow = tn * 16 + l16;
            #pragma unroll
            for (int ki = 0; ki < 2; ki++) {
                Frag kb;
                kb.h[0] = *(const v8bf*)&Ksb[krow * LDST + ki * 32 + lh * 8];
                kb.h[1] = *(const v8bf*)&Ksb[krow * LDST + ki * 32 + lh * 8 + 16];
                s[tn] = wmma_bf16(qa[ki], kb, s[tn]);
            }
        }

        // ---- online softmax (rows live across 16-lane halves)
        float ml[8];
        #pragma unroll
        for (int j = 0; j < 8; j++) {
            float m = -1e30f;
            #pragma unroll
            for (int tn = 0; tn < 4; tn++) {
                s[tn][j] *= 0.25f;                  // SCALE = HEADS^-0.5
                m = fmaxf(m, s[tn][j]);
            }
            ml[j] = m;
        }
        #pragma unroll
        for (int mask = 1; mask <= 8; mask <<= 1)
            #pragma unroll
            for (int j = 0; j < 8; j++)
                ml[j] = fmaxf(ml[j], __shfl_xor(ml[j], mask));

        float alpha[8], ls[8];
        #pragma unroll
        for (int j = 0; j < 8; j++) {
            float nm = fmaxf(rmax[j], ml[j]);
            alpha[j] = __expf(rmax[j] - nm);
            rmax[j] = nm;
            float a2 = 0.f;
            #pragma unroll
            for (int tn = 0; tn < 4; tn++) {
                float p = __expf(s[tn][j] - nm);
                s[tn][j] = p;
                a2 += p;
            }
            ls[j] = a2;
        }
        #pragma unroll
        for (int mask = 1; mask <= 8; mask <<= 1)
            #pragma unroll
            for (int j = 0; j < 8; j++) ls[j] += __shfl_xor(ls[j], mask);
        #pragma unroll
        for (int j = 0; j < 8; j++) rsum[j] = rsum[j] * alpha[j] + ls[j];
        #pragma unroll
        for (int d = 0; d < 4; d++)
            #pragma unroll
            for (int j = 0; j < 8; j++) o[d][j] *= alpha[j];

        // ---- P: C-layout -> per-wave LDS -> A-layout (HW keeps DS in order)
        #pragma unroll
        for (int tn = 0; tn < 4; tn++)
            #pragma unroll
            for (int j = 0; j < 8; j++)
                Pw[(lh * 8 + j) * HD + tn * 16 + l16] = (bf16_t)s[tn][j];
        asm volatile("" ::: "memory");

        Frag pa[2];
        #pragma unroll
        for (int ki = 0; ki < 2; ki++) {
            const bf16_t* pp = Pw + l16 * HD + ki * 32 + lh * 8;
            pa[ki].h[0] = *(const v8bf*)(pp);
            pa[ki].h[1] = *(const v8bf*)(pp + 16);
        }

        // ---- O += P @ V  (Vs rows = head-dim, keys contiguous)
        #pragma unroll
        for (int d = 0; d < 4; d++) {
            int vrow = d * 16 + l16;
            #pragma unroll
            for (int ki = 0; ki < 2; ki++) {
                Frag vb;
                vb.h[0] = *(const v8bf*)&Vsb[vrow * LDST + ki * 32 + lh * 8];
                vb.h[1] = *(const v8bf*)&Vsb[vrow * LDST + ki * 32 + lh * 8 + 16];
                o[d] = wmma_bf16(pa[ki], vb, o[d]);
            }
        }
        asm volatile("" ::: "memory");
    };

    stage_kv(0, 0);                                       // prologue
    for (int kv0 = 0; kv0 < SEQ; kv0 += 128) {
        wait_async0();
        __syncthreads();                                  // buffer 0 ready
        stage_kv(1, kv0 + 64);                            // prefetch next
        kv_body(&Ks[0][0], &Vs[0][0]);

        wait_async0();
        __syncthreads();                                  // buffer 1 ready
        if (kv0 + 128 < SEQ) stage_kv(0, kv0 + 128);
        kv_body(&Ks[1][0], &Vs[1][0]);
    }

    // ---- normalize and store O as bf16 in [B,N,D] row-major
    const int b = bh >> 4, h = bh & 15;
    #pragma unroll
    for (int d = 0; d < 4; d++) {
        int col = h * HD + d * 16 + l16;
        #pragma unroll
        for (int j = 0; j < 8; j++) {
            int q = q0 + lh * 8 + j;
            float val = o[d][j] / rsum[j];
            Ob[(size_t)(b * SEQ + q) * DIM + col] = (bf16_t)val;
        }
    }
}

// ---------------------------------------------------------------- launch
extern "C" void kernel_launch(void* const* d_in, const int* in_sizes, int n_in,
                              void* d_out, int out_size, void* d_ws, size_t ws_size,
                              hipStream_t stream)
{
    const float* x     = (const float*)d_in[0];
    const float* w_qkv = (const float*)d_in[1];
    const float* b_qkv = (const float*)d_in[2];
    const float* w_out = (const float*)d_in[3];
    const float* b_out = (const float*)d_in[4];
    float* out = (float*)d_out;

    char* w = (char*)d_ws;
    bf16_t* Xbf  = (bf16_t*)w;  w += (size_t)MTOT * DIM * 2;
    bf16_t* Wqkv = (bf16_t*)w;  w += (size_t)QKVN * DIM * 2;
    bf16_t* Wout = (bf16_t*)w;  w += (size_t)DIM  * DIM * 2;
    bf16_t* Qb   = (bf16_t*)w;  w += (size_t)MTOT * DIM * 2;
    bf16_t* Kb   = (bf16_t*)w;  w += (size_t)MTOT * DIM * 2;
    bf16_t* Vtb  = (bf16_t*)w;  w += (size_t)MTOT * DIM * 2;
    bf16_t* Ob   = (bf16_t*)w;  w += (size_t)MTOT * DIM * 2;

    const int n1 = MTOT * DIM, n2 = QKVN * DIM, n3 = DIM * DIM;
    cvt_f32_bf16<<<(n1 + 255) / 256, 256, 0, stream>>>(x, Xbf, n1);
    cvt_f32_bf16<<<(n2 + 255) / 256, 256, 0, stream>>>(w_qkv, Wqkv, n2);
    cvt_f32_bf16<<<(n3 + 255) / 256, 256, 0, stream>>>(w_out, Wout, n3);

    gemm_qkv<<<dim3(MTOT / 128, QKVN / 128), 256, 0, stream>>>(
        Xbf, Wqkv, b_qkv, Qb, Kb, Vtb);

    flash_attn<<<dim3(SEQ / 128, BATCH * HEADS), 256, 0, stream>>>(
        Qb, Kb, Vtb, Ob);

    gemm_out<<<dim3(MTOT / 128, DIM / 128), 256, 0, stream>>>(
        Ob, Wout, b_out, out);
}
